// SheafLayer_39195871543807
// MI455X (gfx1250) — matrix operand order repaired and checked by reading, not compiled, not run on hardware
//
#include <hip/hip_runtime.h>

#define STALK 16
#define HID 64

typedef __attribute__((ext_vector_type(16))) _Float16 v16h;
typedef __attribute__((ext_vector_type(8)))  float    v8f;

union V16H { v16h v; _Float16 e[16]; };
union V8F  { v8f  v; float    e[8];  };

// ---------------------------------------------------------------------------
// Pass 1: node degrees (both endpoints of each edge contribute 1)
// ---------------------------------------------------------------------------
__global__ void sheaf_deg_kernel(const long long* __restrict__ ei,
                                 float* __restrict__ deg, int E) {
  int i = blockIdx.x * blockDim.x + threadIdx.x;
  if (i < E) {
    atomicAdd(&deg[(int)ei[i]],     1.0f);
    atomicAdd(&deg[(int)ei[E + i]], 1.0f);
  }
}

// ---------------------------------------------------------------------------
// Pass 2: per-edge MLP (WMMA) + Householder transport + scatter-add messages.
// One wave (32 lanes) handles 16 edges. 8 waves / block.
// ---------------------------------------------------------------------------
__global__ __launch_bounds__(256) void sheaf_edge_kernel(
    const float* __restrict__ x,
    const long long* __restrict__ ei,
    const float* __restrict__ W1, const float* __restrict__ b1,
    const float* __restrict__ W2, const float* __restrict__ b2,
    const float* __restrict__ deg,
    float* __restrict__ acc,
    int E, int numGroups)
{
  // per-wave scratch: hidden tile [16 x 64] f16, v-vectors 2x[16 x 16] f32
  __shared__ _Float16 Hlds[8][16 * HID];
  __shared__ float    Vlds[8][2 * 16 * 16];

  const int tid  = threadIdx.x;
  const int wv   = tid >> 5;
  const int lane = tid & 31;
  const int m    = lane & 15;   // matrix row (A/D M index) / N column
  const int hi   = lane >> 4;   // half-wave select
  const int cb   = hi * 16;     // B-operand K-half base
  const int kb   = hi * 8;      // A-operand K sub-block base
  const int m0   = hi * 8;      // D-layout row base

  _Float16* Hl = Hlds[wv];
  float*    Vl = Vlds[wv];

  const long long* src = ei;
  const long long* dst = ei + E;

  // ---- stage weights into WMMA B-operand register layout (f16) ----
  // B tile layout (16-bit, 32x16): lane<16 holds col N=lane, K={0..7,16..23}
  // packed 2-per-VGPR; lane>=16 holds K={8..15,24..31}.  B1[k][j] = W1[j][k]
  // is two runs of 8 contiguous floats per lane.
  V16H  B1t[4];
  float b1v[4];
#pragma unroll
  for (int n = 0; n < 4; ++n) {
    const float* p = W1 + (size_t)(16 * n + m) * (2 * STALK) + kb;
#pragma unroll
    for (int i = 0; i < 16; ++i) B1t[n].e[i] = (_Float16)p[(i & 7) + ((i & 8) << 1)];
    b1v[n] = b1[16 * n + m];
  }
  V16H B2t[2];
#pragma unroll
  for (int t = 0; t < 2; ++t) {
    const float* p = W2 + (size_t)m * HID + 32 * t + kb;
#pragma unroll
    for (int i = 0; i < 16; ++i) B2t[t].e[i] = (_Float16)p[(i & 7) + ((i & 8) << 1)];
  }
  const float b2v = b2[m];

  const int  group    = blockIdx.x * 8 + wv;
  const int  ee       = group * 16 + m;            // this lane's edge row
  const int  er       = ee < E ? ee : (E - 1);     // clamp tail (uniform flow)
  const bool validRow = (ee < E) && (group < numGroups);

  const long long sN = src[er];
  const long long dN = dst[er];
  const float* xs_p = x + (size_t)sN * STALK + kb;
  const float* xd_p = x + (size_t)dN * STALK + kb;

  // ---- two reflections: pass0 = reflection(xs,xd) -> v_s,
  //                       pass1 = reflection(xd,xs) -> v_d ----
  for (int pass = 0; pass < 2; ++pass) {
    const float* pa = pass ? xd_p : xs_p;   // first 16 cols of concat
    const float* pb = pass ? xs_p : xd_p;   // last 16 cols of concat

    // A-operand (16x32 f16): lane<16: K={0..7,16..23}; lane>=16: {8..15,24..31}
    V16H A;
#pragma unroll
    for (int i = 0; i < 8; ++i) {
      A.e[i]     = (_Float16)pa[i];
      A.e[8 + i] = (_Float16)pb[i];
    }

    // GEMM1: H = relu(A @ W1^T + b1); K=32 -> single WMMA per 16-col tile
#pragma unroll
    for (int n = 0; n < 4; ++n) {
      V8F c{};
      c.v = __builtin_amdgcn_wmma_f32_16x16x32_f16(
          false, A.v, false, B1t[n].v, (short)0, c.v, false, false);
      const int col = 16 * n + m;   // D layout: lane holds column `col`
#pragma unroll
      for (int r = 0; r < 8; ++r) {
        float hv = c.e[r] + b1v[n];
        Hl[(m0 + r) * HID + col] = (_Float16)fmaxf(hv, 0.0f);
      }
    }
    __syncthreads();   // D-layout -> A-layout transpose via LDS

    // GEMM2: V = H @ W2^T + b2; K=64 -> 2 WMMA K-steps
    V8F cv{};
#pragma unroll
    for (int t = 0; t < 2; ++t) {
      V16H A2;
#pragma unroll
      for (int i = 0; i < 8; ++i) {
        A2.e[i]     = Hl[m * HID + 32 * t + kb + i];
        A2.e[8 + i] = Hl[m * HID + 32 * t + 16 + kb + i];
      }
      cv.v = __builtin_amdgcn_wmma_f32_16x16x32_f16(
          false, A2.v, false, B2t[t].v, (short)0, cv.v, false, false);
    }
    float* Vd = Vl + pass * 256;
#pragma unroll
    for (int r = 0; r < 8; ++r) Vd[(m0 + r) * 16 + m] = cv.e[r] + b2v;
    __syncthreads();   // also protects Hl reuse on next pass
  }

  // ---- Householder transport (never materialize the 16x16 matrices) ----
  // lanes 0-15 : edge m, message to src:  +coef * F_s F_d x_d
  // lanes 16-31: edge m, message to dst:  +coef * F_d F_s x_s
  const float* vs = Vl;         // reflection(xs,xd)
  const float* vd = Vl + 256;   // reflection(xd,xs)

  float a1[STALK], a2[STALK], w[STALK];
#pragma unroll
  for (int j = 0; j < STALK; ++j) {
    const float vsj = vs[m * 16 + j];
    const float vdj = vd[m * 16 + j];
    a1[j] = hi ? vsj : vdj;     // reflector applied first
    a2[j] = hi ? vdj : vsj;     // reflector applied second
  }
  const long long wNode = hi ? sN : dN;  // vector being transported
  const long long tNode = hi ? dN : sN;  // accumulation target
  const float* wp = x + (size_t)wNode * STALK;
#pragma unroll
  for (int j = 0; j < STALK; ++j) w[j] = wp[j];

  float n1 = 0.f, n2 = 0.f, d1 = 0.f;
#pragma unroll
  for (int j = 0; j < STALK; ++j) {
    n1 += a1[j] * a1[j];
    n2 += a2[j] * a2[j];
    d1 += a1[j] * w[j];
  }
  const float i1 = 1.0f / fmaxf(sqrtf(n1), 1e-12f);
  const float i2 = 1.0f / fmaxf(sqrtf(n2), 1e-12f);
  const float s1 = 2.0f * d1 * i1 * i1;

  float t[STALK];
  float d2 = 0.f;
#pragma unroll
  for (int j = 0; j < STALK; ++j) {
    t[j] = w[j] - s1 * a1[j];
    d2  += a2[j] * t[j];
  }
  const float s2 = 2.0f * d2 * i2 * i2;

  const float coef = rsqrtf(fmaxf(deg[(int)sN], 1e-5f)) *
                     rsqrtf(fmaxf(deg[(int)dN], 1e-5f));

  if (validRow) {
    float* ap = acc + (size_t)tNode * STALK;
#pragma unroll
    for (int j = 0; j < STALK; ++j)
      atomicAdd(&ap[j], coef * (t[j] - s2 * a2[j]));
  }
}

// ---------------------------------------------------------------------------
// Pass 3: out = relu(x + acc)
// ---------------------------------------------------------------------------
__global__ void sheaf_finalize_kernel(const float* __restrict__ x,
                                      const float* __restrict__ acc,
                                      float* __restrict__ out, int n) {
  int i = blockIdx.x * blockDim.x + threadIdx.x;
  if (i < n) out[i] = fmaxf(x[i] + acc[i], 0.0f);
}

// ---------------------------------------------------------------------------
extern "C" void kernel_launch(void* const* d_in, const int* in_sizes, int n_in,
                              void* d_out, int out_size, void* d_ws, size_t ws_size,
                              hipStream_t stream) {
  const float*     x  = (const float*)d_in[0];
  const long long* ei = (const long long*)d_in[1];
  const float*     W1 = (const float*)d_in[2];
  const float*     b1 = (const float*)d_in[3];
  const float*     W2 = (const float*)d_in[4];
  const float*     b2 = (const float*)d_in[5];
  float* out = (float*)d_out;

  const int N = in_sizes[0] / STALK;
  const int E = in_sizes[1] / 2;

  float* acc = (float*)d_ws;                 // [N*16] f32 message accumulator
  float* deg = acc + (size_t)N * STALK;      // [N]    f32 degrees

  hipMemsetAsync(d_ws, 0, (size_t)(N * STALK + N) * sizeof(float), stream);

  sheaf_deg_kernel<<<(E + 255) / 256, 256, 0, stream>>>(ei, deg, E);

  const int numGroups = (E + 15) / 16;       // 16 edges per wave
  const int blocks    = (numGroups + 7) / 8; // 8 waves per block
  sheaf_edge_kernel<<<blocks, 256, 0, stream>>>(x, ei, W1, b1, W2, b2,
                                                deg, acc, E, numGroups);

  const int tot = N * STALK;
  sheaf_finalize_kernel<<<(tot + 255) / 256, 256, 0, stream>>>(x, acc, out, tot);
}
